// MultiHeadAttention_6150393168162
// MI455X (gfx1250) — compile-verified, matrix operand order checked by reading, not run on hardware
//
#include <hip/hip_runtime.h>
#include <cstdint>

typedef _Float16 h16;
typedef __attribute__((ext_vector_type(16))) _Float16 v16h;
typedef __attribute__((ext_vector_type(8)))  _Float16 v8h;
typedef __attribute__((ext_vector_type(8)))  float    v8f;

#define DD 512
#define SS 4096
#define BB 2
#define HH 8
#define HDIM 64

// workspace layout (units: h16 elements)
constexpr size_t N_XH  = (size_t)BB * SS * DD;   // 4,194,304
constexpr size_t N_W   = (size_t)DD * DD;        //   262,144
constexpr size_t OFF_XH = 0;
constexpr size_t OFF_WP = OFF_XH + N_XH;         // 4 packed weights
constexpr size_t OFF_Q  = OFF_WP + 4 * N_W;      // packed Q fragments
constexpr size_t OFF_K  = OFF_Q + N_XH;          // packed K fragments
constexpr size_t OFF_VT = OFF_K + N_XH;          // packed V fragments
constexpr size_t OFF_AO = OFF_VT + N_XH;         // attention out, row-major f16
// total = 22,020,096 h16 = 44,040,192 bytes

#define QSCALE (0.125f * 1.44269504088896340736f)  // 1/sqrt(64) * log2(e)

// Load 16 halves in the CDNA5 16-bit fragment pattern from row-major memory:
// halves 0..7 from p[0..7], halves 8..15 from p[16..23].
static __device__ __forceinline__ v16h load_pat(const h16* p) {
  v8h lo = *(const v8h*)(p);
  v8h hv = *(const v8h*)(p + 16);
  return __builtin_shufflevector(lo, hv, 0,1,2,3,4,5,6,7,8,9,10,11,12,13,14,15);
}

// 16-lane max reduction via DPP (no LDS): xor1, xor2 (quad_perm),
// row_half_mirror (pairs quads), row_mirror (pairs octets).
template <int CTRL>
static __device__ __forceinline__ float dppmax(float x) {
  int yi = __builtin_amdgcn_update_dpp(0, __builtin_bit_cast(int, x), CTRL, 0xF, 0xF, true);
  return fmaxf(x, __builtin_bit_cast(float, yi));
}
static __device__ __forceinline__ float rowmax16(float x) {
  x = dppmax<0xB1>(x);   // quad_perm [1,0,3,2]  (xor 1)
  x = dppmax<0x4E>(x);   // quad_perm [2,3,0,1]  (xor 2)
  x = dppmax<0x141>(x);  // row_half_mirror      (xor within 8)
  x = dppmax<0x140>(x);  // row_mirror           (xor within 16)
  return x;
}

// fragment (hi, h) from a 0..31 K/contraction index w
static __device__ __forceinline__ void frag_hih(int w, int& hi, int& h) {
  hi = (w >> 3) & 1;
  h  = (w & 7) | ((w & 16) >> 1);
}

// ---------------- x (f32) -> xh (f16) ----------------
__global__ void cvt_x_kernel(const float* __restrict__ x, h16* __restrict__ xh) {
  size_t i = ((size_t)blockIdx.x * 256 + threadIdx.x) * 4;
#pragma unroll
  for (int j = 0; j < 4; ++j) xh[i + j] = (h16)x[i + j];
}

// ---------------- pack W into B-fragment-major f16 layout ----------------
__global__ void pack_w_kernel(const float* __restrict__ Wq, const float* __restrict__ Wk,
                              const float* __restrict__ Wv, const float* __restrict__ Wo,
                              h16* __restrict__ wp) {
  int tid  = blockIdx.x * 256 + threadIdx.x;  // 0..65535
  int w    = tid >> 14;
  int t    = tid & 16383;
  int lane = t & 31;
  int nt   = (t >> 5) & 31;                   // N tile (16 cols)
  int kt   = t >> 10;                         // K tile (32 rows)
  const float* W = (w == 0) ? Wq : (w == 1) ? Wk : (w == 2) ? Wv : Wo;
  int hi = (lane >> 4) & 1;
  int n  = nt * 16 + (lane & 15);
  h16* dst = wp + (size_t)w * N_W + (((size_t)(kt * 32 + nt) * 32 + lane) * 16);
#pragma unroll
  for (int h = 0; h < 16; ++h) {
    int k = ((h < 8) ? h : h + 8) + hi * 8 + kt * 32;
    dst[h] = (h16)W[(size_t)k * DD + n];
  }
}

// ---------------- GEMM: Y[8192,512] = A(f16) * W(packed f16) + bias ----------------
// MODE 0: scatter into packed A/B-fragment layout [bh][tile16][kc][lane][16] (Q and K)
// MODE 2: scatter into packed V-fragment layout  [bh][blk32][t][lane][16]
// MODE 3: store f32 to d_out [B,S,D]
template <int MODE>
__global__ void gemm512_kernel(const h16* __restrict__ A, const h16* __restrict__ Wp,
                               const float* __restrict__ bias, void* __restrict__ dstp,
                               float outScale) {
  int wid  = threadIdx.x >> 5;
  int lane = threadIdx.x & 31;
  int hi   = (lane >> 4) & 1;
  int ln   = lane & 15;
  int work = blockIdx.x * 8 + wid;   // 0..4095
  int mTile = work >> 3;             // 0..511  (16 rows each)
  int nGrp  = work & 7;              // 0..7    (64 cols each)

  v8f acc[4] = {};
  const h16* arow = A + ((size_t)(mTile * 16 + ln)) * DD + hi * 8;

  for (int kt = 0; kt < 16; ++kt) {
    v16h a = load_pat(arow + kt * 32);
#pragma unroll
    for (int t = 0; t < 4; ++t) {
      int nt = nGrp * 4 + t;
      v16h b = *(const v16h*)(Wp + (((size_t)(kt * 32 + nt) * 32 + lane) * 16));
      acc[t] = __builtin_amdgcn_wmma_f32_16x16x32_f16(false, a, false, b,
                                                      (short)0, acc[t], false, false);
    }
  }

#pragma unroll
  for (int t = 0; t < 4; ++t) {
    int nt = nGrp * 4 + t;
    int n  = nt * 16 + ln;
    float bval = bias[n];
#pragma unroll
    for (int r = 0; r < 8; ++r) {
      int m = r + 8 * hi;
      int g = mTile * 16 + m;                 // global row (b*4096 + s)
      float val = (acc[t][r] + bval) * outScale;
      if (MODE == 3) {
        ((float*)dstp)[(size_t)g * DD + n] = val;
      } else {
        int bb = g >> 12, s = g & (SS - 1);
        int hh = n >> 6,  hd = n & 63;
        int bh = bb * HH + hh;
        h16 hvv = (h16)val;
        if (MODE == 0) {
          // A/B-fragment pattern over (row=s within 16-tile, contraction=hd)
          int tb = s >> 4, m16 = s & 15;
          int kc = hd >> 5, w = hd & 31;
          int fhi, fh; frag_hih(w, fhi, fh);
          size_t di = ((((size_t)bh * 256 + tb) * 2 + kc) * 32 + (m16 + fhi * 16)) * 16 + fh;
          ((h16*)dstp)[di] = hvv;
        } else {
          // V fragment: contraction = key (s), n-dim = hd
          int jb = s >> 5, kk = s & 31;
          int fhi, fh; frag_hih(kk, fhi, fh);
          int tt = hd >> 4, l15 = hd & 15;
          size_t di = ((((size_t)bh * 128 + jb) * 4 + tt) * 32 + (l15 + fhi * 16)) * 16 + fh;
          ((h16*)dstp)[di] = hvv;
        }
      }
    }
  }
}

// ---------------- Flash attention ----------------
// 1 wave = 16-query block, 64-key inner blocks; 4 waves/block share (b,h).
// K/V blocks async-staged into double-buffered LDS via global_load_async_to_lds_b128.
// launch_bounds(128,1): allow a full register budget (no spills of Q/score tiles).
__global__ void __launch_bounds__(128, 1)
flash_attn_kernel(const h16* __restrict__ pq, const h16* __restrict__ pk,
                  const h16* __restrict__ pv, h16* __restrict__ AO) {
  __shared__ __align__(16) h16 Pbuf[4][1024];     // per-wave P tile (16x64)
  __shared__ __align__(16) h16 Sbuf[2][8192];     // [buf][ K: 0..4095 | V: 4096..8191 ]
  int tid  = threadIdx.x;
  int wid  = tid >> 5;
  int lane = tid & 31;
  int hi   = (lane >> 4) & 1;
  int ln   = lane & 15;
  int idx  = blockIdx.x * 4 + wid;   // 0..4095
  int qb   = idx & 255;
  int bh   = idx >> 8;               // all 4 waves in a block share bh

  const h16* kbase = pk + (size_t)bh * 512 * 512;   // 256 tiles * 2 kc * 512
  const h16* vbase = pv + (size_t)bh * 512 * 512;   // 128 blks * 4 t  * 512

  const h16* qfr = pq + ((size_t)(bh * 256 + qb) * 2) * 512 + lane * 16;
  v16h aq0 = *(const v16h*)(qfr);          // Q k=0..31  (scaled by 1/8*log2e)
  v16h aq1 = *(const v16h*)(qfr + 512);    // Q k=32..63

  v8f acc[4] = {};
  v8f accl = {};                           // row-sum accumulator (ones-WMMA)
  float mrow[8];
#pragma unroll
  for (int r = 0; r < 8; ++r) mrow[r] = -1e30f;

  v16h bones;
#pragma unroll
  for (int h = 0; h < 16; ++h) bones[h] = (h16)1.0f;

  h16* Pw = &Pbuf[wid][0];

  // stage a 64-key K+V block into Sbuf[buf]: 8192 halves, 128 threads x 8 async b128
  auto stage = [&](int j, int buf) {
    const h16* ks = kbase + (size_t)j * 4096;
    const h16* vs = vbase + (size_t)j * 4096;
#pragma unroll
    for (int i = 0; i < 4; ++i) {
      int c = tid + i * 128;               // 0..511
      uint32_t lk = (uint32_t)(uintptr_t)&Sbuf[buf][c * 8];
      uint32_t lv = (uint32_t)(uintptr_t)&Sbuf[buf][4096 + c * 8];
      asm volatile("global_load_async_to_lds_b128 %0, %1, off"
                   :: "v"(lk), "v"((uint64_t)(uintptr_t)(ks + c * 8)) : "memory");
      asm volatile("global_load_async_to_lds_b128 %0, %1, off"
                   :: "v"(lv), "v"((uint64_t)(uintptr_t)(vs + c * 8)) : "memory");
    }
  };

  stage(0, 0);
  for (int j = 0; j < SS / 64; ++j) {
    int buf = j & 1;
    if (j + 1 < SS / 64) {
      stage(j + 1, buf ^ 1);
      asm volatile("s_wait_asynccnt 0x8" ::: "memory");   // block j's 8 copies done
    } else {
      asm volatile("s_wait_asynccnt 0x0" ::: "memory");
    }
    __syncthreads();

    const h16* kb = &Sbuf[buf][0];
    const h16* vb = &Sbuf[buf][4096];

    // scores: 4 N-tiles (16 keys each) x 2 K-chunks
    v8f s[4];
#pragma unroll
    for (int nt = 0; nt < 4; ++nt) {
      v16h b0 = *(const v16h*)(kb + (nt * 2 + 0) * 512 + lane * 16);
      v16h b1 = *(const v16h*)(kb + (nt * 2 + 1) * 512 + lane * 16);
      v8f c = {};
      c = __builtin_amdgcn_wmma_f32_16x16x32_f16(false, aq0, false, b0, (short)0, c, false, false);
      c = __builtin_amdgcn_wmma_f32_16x16x32_f16(false, aq1, false, b1, (short)0, c, false, false);
      s[nt] = c;
    }

    float alpha[8];
#pragma unroll
    for (int r = 0; r < 8; ++r) {
      float tm = fmaxf(fmaxf(s[0][r], s[1][r]), fmaxf(s[2][r], s[3][r]));
      tm = rowmax16(tm);
      float mn = fmaxf(mrow[r], tm);
      alpha[r] = __builtin_amdgcn_exp2f(mrow[r] - mn);
      mrow[r] = mn;
      int row = r + 8 * hi;
#pragma unroll
      for (int nt = 0; nt < 4; ++nt) {
        float p = __builtin_amdgcn_exp2f(s[nt][r] - mn);
        Pw[row * 64 + nt * 16 + ln] = (h16)p;
      }
    }

    v16h ap0 = load_pat(Pw + ln * 64 + hi * 8);        // P keys 0..31
    v16h ap1 = load_pat(Pw + ln * 64 + 32 + hi * 8);   // P keys 32..63

#pragma unroll
    for (int r = 0; r < 8; ++r) accl[r] *= alpha[r];
    accl = __builtin_amdgcn_wmma_f32_16x16x32_f16(false, ap0, false, bones,
                                                  (short)0, accl, false, false);
    accl = __builtin_amdgcn_wmma_f32_16x16x32_f16(false, ap1, false, bones,
                                                  (short)0, accl, false, false);
#pragma unroll
    for (int t = 0; t < 4; ++t) {
#pragma unroll
      for (int r = 0; r < 8; ++r) acc[t][r] *= alpha[r];
      v16h bv0 = *(const v16h*)(vb + (0 * 4 + t) * 512 + lane * 16);  // keys 0..31
      v16h bv1 = *(const v16h*)(vb + (1 * 4 + t) * 512 + lane * 16);  // keys 32..63
      acc[t] = __builtin_amdgcn_wmma_f32_16x16x32_f16(false, ap0, false, bv0,
                                                      (short)0, acc[t], false, false);
      acc[t] = __builtin_amdgcn_wmma_f32_16x16x32_f16(false, ap1, false, bv1,
                                                      (short)0, acc[t], false, false);
    }
    __syncthreads();   // all waves done with Sbuf[buf] before it is restaged
  }

  int bb = bh >> 3, hh = bh & 7;
#pragma unroll
  for (int t = 0; t < 4; ++t) {
#pragma unroll
    for (int r = 0; r < 8; ++r) {
      int row = r + 8 * hi;
      int sI  = qb * 16 + row;
      int col = hh * 64 + t * 16 + ln;
      float val = acc[t][r] * (1.0f / accl[r]);
      AO[((size_t)(bb * SS + sI)) * DD + col] = (h16)val;
    }
  }
}

extern "C" void kernel_launch(void* const* d_in, const int* in_sizes, int n_in,
                              void* d_out, int out_size, void* d_ws, size_t ws_size,
                              hipStream_t stream) {
  const float* x  = (const float*)d_in[0];
  const float* Wq = (const float*)d_in[1];
  const float* bq = (const float*)d_in[2];
  const float* Wk = (const float*)d_in[3];
  const float* bk = (const float*)d_in[4];
  const float* Wv = (const float*)d_in[5];
  const float* bv = (const float*)d_in[6];
  const float* Wo = (const float*)d_in[7];
  const float* bo = (const float*)d_in[8];

  h16* ws = (h16*)d_ws;
  h16* xh = ws + OFF_XH;
  h16* wp = ws + OFF_WP;
  h16* pq = ws + OFF_Q;
  h16* pk = ws + OFF_K;
  h16* pv = ws + OFF_VT;
  h16* AO = ws + OFF_AO;

  cvt_x_kernel<<<4096, 256, 0, stream>>>(x, xh);
  pack_w_kernel<<<256, 256, 0, stream>>>(Wq, Wk, Wv, Wo, wp);
  gemm512_kernel<0><<<512, 256, 0, stream>>>(xh, wp,           bq, (void*)pq, QSCALE);
  gemm512_kernel<0><<<512, 256, 0, stream>>>(xh, wp + N_W,     bk, (void*)pk, 1.0f);
  gemm512_kernel<2><<<512, 256, 0, stream>>>(xh, wp + 2 * N_W, bv, (void*)pv, 1.0f);
  flash_attn_kernel<<<1024, 128, 0, stream>>>(pq, pk, pv, AO);
  gemm512_kernel<3><<<512, 256, 0, stream>>>(AO, wp + 3 * N_W, bo, d_out, 1.0f);
}
